// Transformer_3255585210401
// MI455X (gfx1250) — matrix operand order repaired and checked by reading, not compiled, not run on hardware
//
#include <hip/hip_runtime.h>
#include <hip/hip_bf16.h>
#include <math.h>

// ---------------- model dims (compile-time) ----------------
#define BSZ 4
#define SEQ 1024
#define DIM 512
#define NH  8
#define HD  512
#define NL  4
#define VOC 32000
#define BT  (BSZ*SEQ)   // 4096 rows
#define HDD (NH*HD)     // 4096
#define LN_EPS 1e-5f

typedef __bf16 bf16_t;
typedef __attribute__((ext_vector_type(16))) __bf16 v16bf;
typedef __attribute__((ext_vector_type(8)))  __bf16 v8bf;
typedef __attribute__((ext_vector_type(8)))  float  v8f;

union V16U { v16bf v; v8bf h[2]; };

// round-to-nearest-even f32 -> bf16 (bit-level)
__device__ __forceinline__ bf16_t to_bf16(float f) {
  unsigned u = __float_as_uint(f);
  u += 0x7FFFu + ((u >> 16) & 1u);
  unsigned short s = (unsigned short)(u >> 16);
  return __builtin_bit_cast(bf16_t, s);
}

// =====================================================================
// Batched bf16 WMMA GEMM:  C[M,N] = A[M,K] * B^T  (B stored [N][K])
// block = 128 threads = 4 waves; each wave computes one 32x64 C tile
// (2 M-subtiles x 4 N-subtiles, 8 f32 accumulators = 64 VGPRs).
// Software-pipelined 2-deep: fragments for step k+32 are loaded before
// the WMMAs of step k issue, so VMEM latency overlaps matrix math.
// grid = (N/64, M/128, nbatch); per-batch offsets: (z/zdiv)*S0 + (z%zdiv)*S1
// MODE: 0 = bf16 store, 1 = f32 store, 2 = f32 + bias,
//       3 = bf16 + bias + relu, 4 = bf16 transposed store (C^T, ldc = rows of C^T)
// Requirements (hold by construction): M % 128 == 0, N % 64 == 0, K % 64 == 0.
// =====================================================================
template<int MODE>
__global__ __launch_bounds__(128)
void gemm_wmma_bf16(const bf16_t* __restrict__ A, const bf16_t* __restrict__ Bm,
                    void* __restrict__ Cv, const float* __restrict__ bias,
                    int K, int lda, int ldb, int ldc,
                    long aS0, long aS1, long bS0, long bS1, long cS0, long cS1,
                    int zdiv)
{
  const int z  = blockIdx.z;
  const int zq = z / zdiv, zr = z % zdiv;
  const size_t aOff = (size_t)zq*aS0 + (size_t)zr*aS1;
  const size_t bOff = (size_t)zq*bS0 + (size_t)zr*bS1;
  const size_t cOff = (size_t)zq*cS0 + (size_t)zr*cS1;

  const int lane = threadIdx.x & 31;
  const int wave = threadIdx.x >> 5;
  const int lq   = lane & 15;      // n (for B/C) or m (for A) within subtile
  const int kh   = lane >> 4;      // K half-select
  const int m0   = (blockIdx.y*4 + wave) * 32;
  const int n0   = blockIdx.x * 64;

  // A 16x32 bf16 fragment: lane half kh holds K = k0 + 8*kh + {0..7} and +16..23
  const bf16_t* arow0 = A + aOff + (size_t)(m0 + lq)*lda + 8*kh;
  const bf16_t* arow1 = arow0 + (size_t)16*lda;
  // B 32x16 fragment: lane half kh holds K = k0 + 8*kh + {0..7} and +16..23 (matches A)
  const bf16_t* b0    = Bm + bOff + (size_t)(n0 + lq)*ldb + 8*kh;

  v8f acc[2][4] = {};
  V16U aC[2], bC[4], aN[2], bN[4];

  auto loadA = [&](V16U (&af)[2], int k0) {
    af[0].h[0] = *reinterpret_cast<const v8bf*>(arow0 + k0);
    af[0].h[1] = *reinterpret_cast<const v8bf*>(arow0 + k0 + 16);
    af[1].h[0] = *reinterpret_cast<const v8bf*>(arow1 + k0);
    af[1].h[1] = *reinterpret_cast<const v8bf*>(arow1 + k0 + 16);
  };
  auto loadB = [&](V16U (&bf)[4], int k0) {
#pragma unroll
    for (int nt = 0; nt < 4; ++nt) {
      const bf16_t* bp = b0 + (size_t)(nt*16)*ldb + k0;
      bf[nt].h[0] = *reinterpret_cast<const v8bf*>(bp);
      bf[nt].h[1] = *reinterpret_cast<const v8bf*>(bp + 16);
    }
  };
  auto mmaStep = [&](V16U (&af)[2], V16U (&bf)[4]) {
#pragma unroll
    for (int mt = 0; mt < 2; ++mt)
#pragma unroll
      for (int nt = 0; nt < 4; ++nt)
        acc[mt][nt] = __builtin_amdgcn_wmma_f32_16x16x32_bf16(
            false, af[mt].v, false, bf[nt].v, (short)0, acc[mt][nt], false, false);
  };

  // 2-deep software pipeline over K (K % 64 == 0, K >= 64)
  loadA(aC, 0); loadB(bC, 0);
#pragma unroll 1
  for (int k0 = 0; k0 < K - 64; k0 += 64) {
    loadA(aN, k0 + 32); loadB(bN, k0 + 32);
    mmaStep(aC, bC);
    loadA(aC, k0 + 64); loadB(bC, k0 + 64);
    mmaStep(aN, bN);
  }
  loadA(aN, K - 32); loadB(bN, K - 32);
  mmaStep(aC, bC);
  mmaStep(aN, bN);

  float*  Cf = reinterpret_cast<float*>(Cv);
  bf16_t* Cb = reinterpret_cast<bf16_t*>(Cv);
#pragma unroll
  for (int nt = 0; nt < 4; ++nt) {
    const int n = n0 + nt*16 + lq;
    float bval = 0.0f;
    if (MODE == 2 || MODE == 3) bval = bias[n];
#pragma unroll
    for (int mt = 0; mt < 2; ++mt) {
      if (MODE == 4) {
        // transposed store: C^T[n, m0+mt*16+8*kh .. +7] packed as one 16B store
        v8bf pk;
#pragma unroll
        for (int r = 0; r < 8; ++r) pk[r] = to_bf16(acc[mt][nt][r]);
        *reinterpret_cast<v8bf*>(Cb + cOff + (size_t)n*ldc +
                                 (size_t)(m0 + mt*16 + 8*kh)) = pk;
      } else {
#pragma unroll
        for (int r = 0; r < 8; ++r) {
          const int m = m0 + mt*16 + 8*kh + r;     // C/D layout: VGPR r, half kh
          const size_t o = cOff + (size_t)m*ldc + (size_t)n;
          float v = acc[mt][nt][r] + bval;
          if (MODE == 3) v = fmaxf(v, 0.0f);
          if (MODE == 1 || MODE == 2) Cf[o] = v;
          else                         Cb[o] = to_bf16(v);
        }
      }
    }
  }
}

// ---------------- embedding: x = tok_emb[idx] + pos_emb ----------------
__global__ __launch_bounds__(256)
void embed_kernel(const int* __restrict__ idx, const float* __restrict__ tok,
                  const float* __restrict__ pos, float* __restrict__ x,
                  bf16_t* __restrict__ xb)
{
  long gid = (long)blockIdx.x*256 + threadIdx.x;
  if (gid >= (long)BT*DIM) return;
  int row = (int)(gid / DIM);
  int d   = (int)(gid % DIM);
  int t   = row % SEQ;
  float v = tok[(size_t)idx[row]*DIM + d] + pos[(size_t)t*DIM + d];
  x[gid]  = v;
  xb[gid] = to_bf16(v);
}

// ---------------- f32 -> bf16 transpose convert: [b][R][C] -> [b][C][R] ----------------
__global__ __launch_bounds__(256)
void convT_kernel(const float* __restrict__ src, bf16_t* __restrict__ dst,
                  int R, int C, long total)
{
  long gid = (long)blockIdx.x*256 + threadIdx.x;
  if (gid >= total) return;
  long rc  = (long)R * C;
  long bi  = gid / rc;
  long rem = gid - bi*rc;
  int r = (int)(rem / C), c = (int)(rem % C);
  dst[bi*rc + (long)c*R + r] = to_bf16(src[gid]);
}

// ---------------- row softmax (T=1024): attn = softmax(scores * scale) -> bf16 ----------------
__global__ __launch_bounds__(256)
void softmax_kernel(const float* __restrict__ S, bf16_t* __restrict__ P, float scale)
{
  static_assert(SEQ == 1024, "softmax assumes T=1024");
  const int tid = threadIdx.x;
  const size_t base = (size_t)blockIdx.x * SEQ;
  __shared__ float red[256];
  float vals[4];
  float mx = -INFINITY;
#pragma unroll
  for (int j = 0; j < 4; ++j) {
    vals[j] = S[base + j*256 + tid] * scale;
    mx = fmaxf(mx, vals[j]);
  }
  red[tid] = mx; __syncthreads();
  for (int s2 = 128; s2 > 0; s2 >>= 1) {
    if (tid < s2) red[tid] = fmaxf(red[tid], red[tid + s2]);
    __syncthreads();
  }
  mx = red[0]; __syncthreads();
  float sum = 0.0f;
#pragma unroll
  for (int j = 0; j < 4; ++j) { vals[j] = __expf(vals[j] - mx); sum += vals[j]; }
  red[tid] = sum; __syncthreads();
  for (int s2 = 128; s2 > 0; s2 >>= 1) {
    if (tid < s2) red[tid] += red[tid + s2];
    __syncthreads();
  }
  const float inv = 1.0f / red[0];
#pragma unroll
  for (int j = 0; j < 4; ++j) P[base + j*256 + tid] = to_bf16(vals[j] * inv);
}

// ---------------- fused residual add + LayerNorm (D=512) ----------------
// out = LN(a + resid); writes bf16 always, f32 optionally (outF may alias resid)
__global__ __launch_bounds__(256)
void add_ln_kernel(const float* __restrict__ a, const float* __restrict__ resid,
                   const float* __restrict__ g, const float* __restrict__ be,
                   float* __restrict__ outF, bf16_t* __restrict__ outB)
{
  static_assert(DIM == 512, "LN assumes D=512");
  const int tid = threadIdx.x;
  const size_t base = (size_t)blockIdx.x * DIM;
  float v0 = a[base + tid      ] + resid[base + tid      ];
  float v1 = a[base + tid + 256] + resid[base + tid + 256];
  __shared__ float rs[256], rq[256];
  rs[tid] = v0 + v1;
  rq[tid] = v0*v0 + v1*v1;
  __syncthreads();
  for (int s2 = 128; s2 > 0; s2 >>= 1) {
    if (tid < s2) { rs[tid] += rs[tid + s2]; rq[tid] += rq[tid + s2]; }
    __syncthreads();
  }
  const float mean = rs[0] * (1.0f / DIM);
  const float var  = rq[0] * (1.0f / DIM) - mean*mean;
  const float rstd = rsqrtf(var + LN_EPS);
  float y0 = (v0 - mean) * rstd * g[tid      ] + be[tid      ];
  float y1 = (v1 - mean) * rstd * g[tid + 256] + be[tid + 256];
  if (outF) { outF[base + tid] = y0; outF[base + tid + 256] = y1; }
  outB[base + tid      ] = to_bf16(y0);
  outB[base + tid + 256] = to_bf16(y1);
}

// =====================================================================
extern "C" void kernel_launch(void* const* d_in, const int* in_sizes, int n_in,
                              void* d_out, int out_size, void* d_ws, size_t ws_size,
                              hipStream_t stream)
{
  (void)in_sizes; (void)n_in; (void)out_size; (void)ws_size;

  const int*   idx  = (const int*)  d_in[0];
  const float* tok  = (const float*)d_in[1];
  const float* pos  = (const float*)d_in[2];
  const float* Wq   = (const float*)d_in[3];
  const float* Wk   = (const float*)d_in[4];
  const float* Wv   = (const float*)d_in[5];
  const float* Wo   = (const float*)d_in[6];
  const float* bo   = (const float*)d_in[7];
  const float* ln1g = (const float*)d_in[8];
  const float* ln1b = (const float*)d_in[9];
  const float* ln2g = (const float*)d_in[10];
  const float* ln2b = (const float*)d_in[11];
  const float* W1   = (const float*)d_in[12];
  const float* b1   = (const float*)d_in[13];
  const float* W2   = (const float*)d_in[14];
  const float* b2   = (const float*)d_in[15];
  const float* lm_w = (const float*)d_in[16];
  const float* lm_b = (const float*)d_in[17];
  float* out = (float*)d_out;

  // ---- workspace bump allocator (all offsets 256B aligned) ----
  char* wsb = (char*)d_ws;
  size_t cur = 0;
  auto balloc = [&](size_t bytes) -> void* {
    void* p = wsb + cur;
    cur = (cur + bytes + 255) & ~(size_t)255;
    return p;
  };
  // bf16 transposed weights ([N][K] layout for the B operand)
  bf16_t* wqT = (bf16_t*)balloc((size_t)NL*NH*HD*DIM*2);
  bf16_t* wkT = (bf16_t*)balloc((size_t)NL*NH*HD*DIM*2);
  bf16_t* wvT = (bf16_t*)balloc((size_t)NL*NH*HD*DIM*2);
  bf16_t* woT = (bf16_t*)balloc((size_t)NL*DIM*HDD*2);
  bf16_t* w1T = (bf16_t*)balloc((size_t)NL*DIM*DIM*2);
  bf16_t* w2T = (bf16_t*)balloc((size_t)NL*DIM*DIM*2);
  bf16_t* lmT = (bf16_t*)balloc((size_t)VOC*DIM*2);
  // activations
  float*  x   = (float*) balloc((size_t)BT*DIM*4);
  bf16_t* xb  = (bf16_t*)balloc((size_t)BT*DIM*2);
  bf16_t* qb  = (bf16_t*)balloc((size_t)BSZ*NH*SEQ*HD*2);
  bf16_t* kb  = (bf16_t*)balloc((size_t)BSZ*NH*SEQ*HD*2);
  bf16_t* vTb = (bf16_t*)balloc((size_t)BSZ*NH*HD*SEQ*2);   // V^T: [b,h,e,t]
  float*  sc  = (float*) balloc((size_t)BSZ*NH*SEQ*SEQ*4);
  bf16_t* at  = (bf16_t*)balloc((size_t)BSZ*NH*SEQ*SEQ*2);
  bf16_t* ob  = (bf16_t*)balloc((size_t)BT*HDD*2);          // concat heads [b,t,h*e]
  float*  ao  = (float*) balloc((size_t)BT*DIM*4);
  bf16_t* hb  = (bf16_t*)balloc((size_t)BT*DIM*2);
  bf16_t* fh  = (bf16_t*)balloc((size_t)BT*DIM*2);
  float*  ff  = (float*) balloc((size_t)BT*DIM*4);

  // ---- weight convert + transpose (bf16, [out][in]) ----
  {
    long tot = (long)NL*NH*DIM*HD;
    int nb = (int)((tot + 255)/256);
    convT_kernel<<<nb,256,0,stream>>>(Wq, wqT, DIM, HD, tot);
    convT_kernel<<<nb,256,0,stream>>>(Wk, wkT, DIM, HD, tot);
    convT_kernel<<<nb,256,0,stream>>>(Wv, wvT, DIM, HD, tot);
  }
  { long tot = (long)NL*HDD*DIM;
    convT_kernel<<<(int)((tot+255)/256),256,0,stream>>>(Wo, woT, HDD, DIM, tot); }
  { long tot = (long)NL*DIM*DIM;
    int nb = (int)((tot+255)/256);
    convT_kernel<<<nb,256,0,stream>>>(W1, w1T, DIM, DIM, tot);
    convT_kernel<<<nb,256,0,stream>>>(W2, w2T, DIM, DIM, tot); }
  { long tot = (long)DIM*VOC;
    convT_kernel<<<(int)((tot+255)/256),256,0,stream>>>(lm_w, lmT, DIM, VOC, tot); }

  // ---- embedding ----
  { long tot = (long)BT*DIM;
    embed_kernel<<<(int)((tot+255)/256),256,0,stream>>>(idx, tok, pos, x, xb); }

  const float scale = 0.044194173824159216f;   // 1/sqrt(512)

  for (int l = 0; l < NL; ++l) {
    const bf16_t* wq_l = wqT + (size_t)l*NH*HD*DIM;
    const bf16_t* wk_l = wkT + (size_t)l*NH*HD*DIM;
    const bf16_t* wv_l = wvT + (size_t)l*NH*HD*DIM;
    const bf16_t* wo_l = woT + (size_t)l*DIM*HDD;
    const bf16_t* w1_l = w1T + (size_t)l*DIM*DIM;
    const bf16_t* w2_l = w2T + (size_t)l*DIM*DIM;

    // Q,K: [b,h,t,e] = x[b] @ Wq[l,h]^T ; z = b*H + h
    gemm_wmma_bf16<0><<<dim3(HD/64, SEQ/128, BSZ*NH), 128, 0, stream>>>(
        xb, wq_l, qb, nullptr, DIM, DIM, DIM, HD,
        (long)SEQ*DIM, 0, 0, (long)HD*DIM,
        (long)NH*SEQ*HD, (long)SEQ*HD, NH);
    gemm_wmma_bf16<0><<<dim3(HD/64, SEQ/128, BSZ*NH), 128, 0, stream>>>(
        xb, wk_l, kb, nullptr, DIM, DIM, DIM, HD,
        (long)SEQ*DIM, 0, 0, (long)HD*DIM,
        (long)NH*SEQ*HD, (long)SEQ*HD, NH);
    // V stored transposed [b,h,e,t] so attn@V consumes it as B operand
    gemm_wmma_bf16<4><<<dim3(HD/64, SEQ/128, BSZ*NH), 128, 0, stream>>>(
        xb, wv_l, vTb, nullptr, DIM, DIM, DIM, /*ldc=T*/SEQ,
        (long)SEQ*DIM, 0, 0, (long)HD*DIM,
        (long)NH*HD*SEQ, (long)HD*SEQ, NH);

    // scores[b,h] = q @ k^T (k rows already = [N][K] layout)
    gemm_wmma_bf16<1><<<dim3(SEQ/64, SEQ/128, BSZ*NH), 128, 0, stream>>>(
        qb, kb, sc, nullptr, HD, HD, HD, SEQ,
        (long)SEQ*HD, 0, (long)SEQ*HD, 0, (long)SEQ*SEQ, 0, 1);

    softmax_kernel<<<BSZ*NH*SEQ, 256, 0, stream>>>(sc, at, scale);

    // o[b,t,h*e] = attn @ v  (B operand = V^T rows, ldb = T)
    gemm_wmma_bf16<0><<<dim3(HD/64, SEQ/128, BSZ*NH), 128, 0, stream>>>(
        at, vTb, ob, nullptr, SEQ, SEQ, SEQ, HDD,
        (long)NH*SEQ*SEQ, (long)SEQ*SEQ,
        (long)NH*HD*SEQ,  (long)HD*SEQ,
        (long)SEQ*HDD,    (long)HD, NH);

    // attn_out = o @ Wo[l] + bo[l]
    gemm_wmma_bf16<2><<<dim3(DIM/64, BT/128, 1), 128, 0, stream>>>(
        ob, wo_l, ao, bo + (size_t)l*DIM, HDD, HDD, HDD, DIM,
        0, 0, 0, 0, 0, 0, 1);

    // h = LN(attn_out + x) -> bf16 only
    add_ln_kernel<<<BT, 256, 0, stream>>>(
        ao, x, ln1g + (size_t)l*DIM, ln1b + (size_t)l*DIM, nullptr, hb);

    // ffh = relu(h @ W1 + b1) -> bf16
    gemm_wmma_bf16<3><<<dim3(DIM/64, BT/128, 1), 128, 0, stream>>>(
        hb, w1_l, fh, b1 + (size_t)l*DIM, DIM, DIM, DIM, DIM,
        0, 0, 0, 0, 0, 0, 1);

    // ff = ffh @ W2 + b2 -> f32
    gemm_wmma_bf16<2><<<dim3(DIM/64, BT/128, 1), 128, 0, stream>>>(
        fh, w2_l, ff, b2 + (size_t)l*DIM, DIM, DIM, DIM, DIM,
        0, 0, 0, 0, 0, 0, 1);

    // x = LN(ff + x)  (second residual uses block input x, per reference)
    add_ln_kernel<<<BT, 256, 0, stream>>>(
        ff, x, ln2g + (size_t)l*DIM, ln2b + (size_t)l*DIM, x, xb);
  }

  // logits = x @ lm_w + lm_b  -> f32 [B,T,V]
  gemm_wmma_bf16<2><<<dim3(VOC/64, BT/128, 1), 128, 0, stream>>>(
      xb, lmT, out, lm_b, DIM, DIM, DIM, VOC,
      0, 0, 0, 0, 0, 0, 1);
}
